// Phi3VModel_28544352650004
// MI455X (gfx1250) — matrix operand order repaired and checked by reading, not compiled, not run on hardware
//
#include <hip/hip_runtime.h>
#include <hip/hip_bf16.h>
#include <math.h>

// ---------------------------------------------------------------------------
// Model constants (from reference)
// ---------------------------------------------------------------------------
#define LSEQ 2048
#define DMODEL 3072
#define NHEAD 32
#define HDIM 96
#define IFF 8192
#define NLAYER 2
#define QKVN (3 * NHEAD * HDIM)   // 9216
#define OHH  (NHEAD * HDIM)       // 3072

typedef __attribute__((ext_vector_type(16))) _Float16 v16h;
typedef __attribute__((ext_vector_type(8)))  float    v8f;
typedef __attribute__((ext_vector_type(4)))  unsigned int u32x4;
typedef __attribute__((ext_vector_type(8)))  int i32x8;
typedef __attribute__((ext_vector_type(4)))  int i32x4;

// ---------------------------------------------------------------------------
// Tensor Data Mover: 2D tile load Global -> LDS (ISA cdna5 ch.7/8, D# layout)
// group0: [1:0]=count=1, [63:32]=lds_addr, [120:64]=global_addr, [127:126]=2
// group1: [17:16]=data_size, [20]=pad_en, [24:22]=pad_interval, [31:25]=pad_amt
//         [79:48]=tensor_dim0, [111:80]=tensor_dim1, [127:112]=tile_dim0,
//         [143:128]=tile_dim1, [207:160]=tensor_dim0_stride
// pad_amount=0 -> 1 DWORD of LDS padding after every 2^(interval+1) DWORDs.
// ---------------------------------------------------------------------------
__device__ static inline void tdm_load_2d(unsigned lds_off, const void* gptr,
                                          unsigned data_size_code,  // 0=1B,1=2B,2=4B
                                          unsigned tile_d0, unsigned tile_d1,
                                          unsigned long long d0_stride_elems,
                                          unsigned pad_interval, bool pad_en) {
  unsigned long long ga = (unsigned long long)(uintptr_t)gptr;
  u32x4 g0;
  g0[0] = 1u;                                     // count=1 (valid), user mode
  g0[1] = lds_off;                                // lds_addr (bytes)
  g0[2] = (unsigned)(ga & 0xFFFFFFFFull);         // global_addr[31:0]
  g0[3] = (unsigned)((ga >> 32) & 0x01FFFFFFull)  // global_addr[56:32]
          | (2u << 30);                           // type = 2 ("image")
  i32x8 g1;
  g1[0] = (int)((data_size_code << 16) | (pad_en ? (1u << 20) : 0u) |
                ((pad_interval & 7u) << 22));     // pad_amount=0 (1 DWORD)
  g1[1] = (int)((tile_d0 & 0xFFFFu) << 16);       // tensor_dim0[15:0] (=tile)
  g1[2] = (int)(((tile_d0 >> 16) & 0xFFFFu) |     // tensor_dim0[31:16]
                ((tile_d1 & 0xFFFFu) << 16));     // tensor_dim1[15:0] (=tile)
  g1[3] = (int)(((tile_d1 >> 16) & 0xFFFFu) |     // tensor_dim1[31:16]
                ((tile_d0 & 0xFFFFu) << 16));     // tile_dim0
  g1[4] = (int)(tile_d1 & 0xFFFFu);               // tile_dim1; tile_dim2 = 0
  g1[5] = (int)(unsigned)(d0_stride_elems & 0xFFFFFFFFull);
  g1[6] = (int)(unsigned)((d0_stride_elems >> 32) & 0xFFFFull);
  g1[7] = 0;                                      // tensor_dim1_stride = 0 (2D)
  i32x4 zz = {0, 0, 0, 0};
#if __clang_major__ >= 23
  i32x8 z8 = {0, 0, 0, 0, 0, 0, 0, 0};
  __builtin_amdgcn_tensor_load_to_lds(g0, g1, zz, zz, z8, 0);
#else
  __builtin_amdgcn_tensor_load_to_lds(g0, g1, zz, zz, 0);
#endif
}

// ---------------------------------------------------------------------------
// Elementwise: h = x
// ---------------------------------------------------------------------------
__global__ void __launch_bounds__(256) copy_f32(const float* __restrict__ src,
                                                float* __restrict__ dst, int n) {
  int i = blockIdx.x * 256 + threadIdx.x;
  if (i < n) dst[i] = src[i];
}

// ---------------------------------------------------------------------------
// RMSNorm: one block per row. y16 = (w * h * rsqrt(mean(h^2)+eps)) as f16
// ---------------------------------------------------------------------------
__global__ void __launch_bounds__(256) rmsnorm_f16(const float* __restrict__ h,
                                                   const float* __restrict__ w,
                                                   _Float16* __restrict__ y16) {
  __shared__ float red[256];
  const int row = blockIdx.x;
  const int tid = threadIdx.x;
  const float* hr = h + (size_t)row * DMODEL;
  float s = 0.f;
  for (int j = tid; j < DMODEL; j += 256) { float v = hr[j]; s += v * v; }
  red[tid] = s;
  __syncthreads();
  for (int st = 128; st > 0; st >>= 1) {
    if (tid < st) red[tid] += red[tid + st];
    __syncthreads();
  }
  const float inv = rsqrtf(red[0] / (float)DMODEL + 1e-5f);
  _Float16* yr = y16 + (size_t)row * DMODEL;
  for (int j = tid; j < DMODEL; j += 256)
    yr[j] = (_Float16)(w[j] * hr[j] * inv);
}

// ---------------------------------------------------------------------------
// WMMA GEMM:  C[M,N] (fp32) = A16[M,K] (f16) * W[K,N] (fp32 -> f16 in frag)
// EPI==0: store, EPI==1: accumulate (residual add)
// Block 256 thr (8 waves). Tile 64(M) x 128(N), K-step 32. Wave = 32x32 out.
// Tiles staged by the Tensor Data Mover, double buffered, one issuing wave;
// TDM pad feature gives conflict-free LDS strides (A: 34 f16, B: 129 f32).
// blockIdx.x = M-tile so launch order keeps one weight panel hot in L2.
// ---------------------------------------------------------------------------
#define BM 64
#define BN 128
#define BK 32
#define LDA_S 34     // 32 f16 + 1 DWORD TDM pad
#define LDB_S 129    // 128 f32 + 1 DWORD TDM pad

template <int EPI>
__global__ void __launch_bounds__(256)
gemm_wmma(const _Float16* __restrict__ A, const float* __restrict__ W,
          float* __restrict__ C, int M, int N, int K) {
  __shared__ _Float16 As2[2][BM * LDA_S];
  __shared__ float    Bs2[2][BK * LDB_S];

  const int tid  = threadIdx.x;
  const int lane = tid & 31;
  const int wave = tid >> 5;
  const int half = lane >> 4;
  const int l16  = lane & 15;
  const int m0 = blockIdx.x * BM;     // M fastest in launch order
  const int n0 = blockIdx.y * BN;
  const int wm = wave >> 2;   // 0..1
  const int wn = wave & 3;    // 0..3
  const bool issuer = (wave == 0);

  v8f acc[2][2];
#pragma unroll
  for (int a = 0; a < 2; ++a)
#pragma unroll
    for (int b = 0; b < 2; ++b)
      acc[a][b] = (v8f){0.f, 0.f, 0.f, 0.f, 0.f, 0.f, 0.f, 0.f};

  // ---- prologue: TDM-load tile kt=0 into buffer 0 ----
  if (issuer) {
    tdm_load_2d((unsigned)(uintptr_t)&As2[0][0], A + (size_t)m0 * K,
                /*f16*/1, BK, BM, (unsigned long long)K, /*16 DW rows*/3, true);
    tdm_load_2d((unsigned)(uintptr_t)&Bs2[0][0], W + (size_t)0 * N + n0,
                /*f32*/2, BN, BK, (unsigned long long)N, /*128 DW rows*/6, true);
    __builtin_amdgcn_s_wait_tensorcnt(0);
  }
  __syncthreads();

  int cur = 0;
  for (int kt = 0; kt < K; kt += BK) {
    // ---- issue next tile into the other buffer (overlaps with WMMA) ----
    if (issuer && kt + BK < K) {
      tdm_load_2d((unsigned)(uintptr_t)&As2[cur ^ 1][0],
                  A + (size_t)m0 * K + (kt + BK),
                  1, BK, BM, (unsigned long long)K, 3, true);
      tdm_load_2d((unsigned)(uintptr_t)&Bs2[cur ^ 1][0],
                  W + (size_t)(kt + BK) * N + n0,
                  2, BN, BK, (unsigned long long)N, 6, true);
    }

    // A fragments: lane holds row (lane&15); elems 0..7 -> K=8*half+e,
    // elems 8..15 -> K=16+8*half+e   (ISA 7.12.2, 16-bit A 16x32)
    v16h af[2];
#pragma unroll
    for (int a = 0; a < 2; ++a) {
      const _Float16* p = &As2[cur][(wm * 32 + a * 16 + l16) * LDA_S];
#pragma unroll
      for (int e = 0; e < 8; ++e) {
        af[a][e]     = p[8 * half + e];
        af[a][e + 8] = p[16 + 8 * half + e];
      }
    }
    // B fragments: lane holds col (lane&15); elem r -> K = 16*(r>>3) + 8*half + (r&7)
    v16h bf[2];
#pragma unroll
    for (int b = 0; b < 2; ++b) {
      const int col = wn * 32 + b * 16 + l16;
#pragma unroll
      for (int r = 0; r < 16; ++r)
        bf[b][r] = (_Float16)Bs2[cur][(16 * (r >> 3) + 8 * half + (r & 7)) * LDB_S + col];
    }
#pragma unroll
    for (int a = 0; a < 2; ++a)
#pragma unroll
      for (int b = 0; b < 2; ++b)
        acc[a][b] = __builtin_amdgcn_wmma_f32_16x16x32_f16(
            false, af[a], false, bf[b], (short)0, acc[a][b], false, false);

    // ---- TDM completion for next buffer, then swap ----
    if (issuer && kt + BK < K) __builtin_amdgcn_s_wait_tensorcnt(0);
    __syncthreads();
    cur ^= 1;
  }

  // D layout: vgpr j -> row j + 8*half; col = lane&15
#pragma unroll
  for (int a = 0; a < 2; ++a)
#pragma unroll
    for (int b = 0; b < 2; ++b)
#pragma unroll
      for (int j = 0; j < 8; ++j) {
        const int m = m0 + wm * 32 + a * 16 + j + 8 * half;
        const int n = n0 + wn * 32 + b * 16 + l16;
        const size_t off = (size_t)m * N + n;
        if (EPI == 0) C[off] = acc[a][b][j];
        else          C[off] += acc[a][b][j];
      }
}

// ---------------------------------------------------------------------------
// RoPE + QKV split:  qkv fp32 [L, 9216] -> q16/k16/v16 f16 [H, L, HD]
// ---------------------------------------------------------------------------
__global__ void __launch_bounds__(256)
rope_split(const float* __restrict__ qkv, const float* __restrict__ cs,
           const float* __restrict__ sn, _Float16* __restrict__ q16,
           _Float16* __restrict__ k16, _Float16* __restrict__ v16) {
  const int idx = blockIdx.x * 256 + threadIdx.x;
  const int total = LSEQ * NHEAD * (HDIM / 2);
  if (idx >= total) return;
  const int l  = idx / (NHEAD * (HDIM / 2));
  const int r  = idx - l * (NHEAD * (HDIM / 2));
  const int hh = r / (HDIM / 2);
  const int d  = r - hh * (HDIM / 2);

  const size_t row = (size_t)l * QKVN + hh * HDIM;
  const float c1 = cs[(size_t)l * HDIM + d];
  const float s1 = sn[(size_t)l * HDIM + d];
  const float c2 = cs[(size_t)l * HDIM + d + HDIM / 2];
  const float s2 = sn[(size_t)l * HDIM + d + HDIM / 2];

  const float q1 = qkv[row + d], q2 = qkv[row + d + HDIM / 2];
  const float k1 = qkv[row + OHH + d], k2 = qkv[row + OHH + d + HDIM / 2];
  const float v1 = qkv[row + 2 * OHH + d], v2 = qkv[row + 2 * OHH + d + HDIM / 2];

  const size_t ob = (size_t)hh * LSEQ * HDIM + (size_t)l * HDIM;
  q16[ob + d]            = (_Float16)(q1 * c1 - q2 * s1);
  q16[ob + d + HDIM / 2] = (_Float16)(q2 * c2 + q1 * s2);
  k16[ob + d]            = (_Float16)(k1 * c1 - k2 * s1);
  k16[ob + d + HDIM / 2] = (_Float16)(k2 * c2 + k1 * s2);
  v16[ob + d]            = (_Float16)v1;
  v16[ob + d + HDIM / 2] = (_Float16)v2;
}

// ---------------------------------------------------------------------------
// Flash attention: one wave per (head, 16-query tile). Causal.
// ---------------------------------------------------------------------------
__global__ void __launch_bounds__(32)
attn_flash(const _Float16* __restrict__ q, const _Float16* __restrict__ k,
           const _Float16* __restrict__ v, _Float16* __restrict__ o16) {
  __shared__ _Float16 Ps[16 * 32];
  const int lane = threadIdx.x & 31;
  const int half = lane >> 4;
  const int l16  = lane & 15;
  const int head = blockIdx.y;
  const int q0   = blockIdx.x * 16;

  const _Float16* qh = q + (size_t)head * LSEQ * HDIM;
  const _Float16* kh = k + (size_t)head * LSEQ * HDIM;
  const _Float16* vh = v + (size_t)head * LSEQ * HDIM;

  v16h qf[3];
#pragma unroll
  for (int c = 0; c < 3; ++c) {
    const _Float16* p = qh + (size_t)(q0 + l16) * HDIM + c * 32;
#pragma unroll
    for (int e = 0; e < 8; ++e) {
      qf[c][e]     = p[8 * half + e];
      qf[c][e + 8] = p[16 + 8 * half + e];
    }
  }

  float mrow[8], lrow[8];
  v8f oacc[6];
#pragma unroll
  for (int j = 0; j < 8; ++j) { mrow[j] = -1e30f; lrow[j] = 0.f; }
#pragma unroll
  for (int nt = 0; nt < 6; ++nt)
    oacc[nt] = (v8f){0.f, 0.f, 0.f, 0.f, 0.f, 0.f, 0.f, 0.f};

  const float scale = rsqrtf((float)HDIM);
  const int kend = q0 + 16;

  for (int kb = 0; kb < kend; kb += 32) {
    v8f s[2];
#pragma unroll
    for (int t = 0; t < 2; ++t) {
      const _Float16* kp = kh + (size_t)(kb + t * 16 + l16) * HDIM;
      v8f st = (v8f){0.f, 0.f, 0.f, 0.f, 0.f, 0.f, 0.f, 0.f};
#pragma unroll
      for (int c = 0; c < 3; ++c) {
        v16h kf;
#pragma unroll
        for (int r = 0; r < 16; ++r) kf[r] = kp[c * 32 + 16 * (r >> 3) + 8 * half + (r & 7)];
        st = __builtin_amdgcn_wmma_f32_16x16x32_f16(
            false, qf[c], false, kf, (short)0, st, false, false);
      }
      s[t] = st;
    }
    float rmax[8];
#pragma unroll
    for (int j = 0; j < 8; ++j) {
      const int qrow = q0 + j + 8 * half;
      float v0 = s[0][j] * scale;
      float v1 = s[1][j] * scale;
      if (kb + l16 > qrow)      v0 = -1e9f;
      if (kb + 16 + l16 > qrow) v1 = -1e9f;
      s[0][j] = v0; s[1][j] = v1;
      float mx = fmaxf(v0, v1);
      for (int d = 1; d < 16; d <<= 1) mx = fmaxf(mx, __shfl_xor(mx, d, 32));
      rmax[j] = mx;
    }
    float alpha[8];
#pragma unroll
    for (int j = 0; j < 8; ++j) {
      const float mnew = fmaxf(mrow[j], rmax[j]);
      alpha[j] = __expf(mrow[j] - mnew);
      mrow[j] = mnew;
    }
#pragma unroll
    for (int j = 0; j < 8; ++j) {
      const float p0 = __expf(s[0][j] - mrow[j]);
      const float p1 = __expf(s[1][j] - mrow[j]);
      const int row = j + 8 * half;
      Ps[row * 32 + l16]      = (_Float16)p0;
      Ps[row * 32 + 16 + l16] = (_Float16)p1;
      float rs = p0 + p1;
      for (int d = 1; d < 16; d <<= 1) rs += __shfl_xor(rs, d, 32);
      lrow[j] = lrow[j] * alpha[j] + rs;
    }
    __syncthreads();
    v16h pf;
    {
      const _Float16* pp = Ps + l16 * 32;
#pragma unroll
      for (int e = 0; e < 8; ++e) {
        pf[e]     = pp[8 * half + e];
        pf[e + 8] = pp[16 + 8 * half + e];
      }
    }
#pragma unroll
    for (int nt = 0; nt < 6; ++nt) {
      v8f t = oacc[nt];
#pragma unroll
      for (int j = 0; j < 8; ++j) t[j] *= alpha[j];
      v16h vf;
#pragma unroll
      for (int r = 0; r < 16; ++r)
        vf[r] = vh[(size_t)(kb + 16 * (r >> 3) + 8 * half + (r & 7)) * HDIM + nt * 16 + l16];
      oacc[nt] = __builtin_amdgcn_wmma_f32_16x16x32_f16(
          false, pf, false, vf, (short)0, t, false, false);
    }
    __syncthreads();
  }

#pragma unroll
  for (int nt = 0; nt < 6; ++nt)
#pragma unroll
    for (int j = 0; j < 8; ++j) {
      const int row = q0 + j + 8 * half;
      o16[(size_t)row * OHH + head * HDIM + nt * 16 + l16] =
          (_Float16)(oacc[nt][j] / lrow[j]);
    }
}

// ---------------------------------------------------------------------------
// SwiGLU gate: gu fp32 [L, 2I] -> act16 f16 [L, I];  silu(g)*u
// ---------------------------------------------------------------------------
__global__ void __launch_bounds__(256)
silu_gate(const float* __restrict__ gu, _Float16* __restrict__ act16) {
  const int idx = blockIdx.x * 256 + threadIdx.x;
  if (idx >= LSEQ * IFF) return;
  const int l = idx >> 13;          // / IFF
  const int i = idx & (IFF - 1);
  const float g = gu[(size_t)l * (2 * IFF) + i];
  const float u = gu[(size_t)l * (2 * IFF) + IFF + i];
  act16[idx] = (_Float16)((g / (1.f + __expf(-g))) * u);
}

// ---------------------------------------------------------------------------
// Host orchestration
// ---------------------------------------------------------------------------
extern "C" void kernel_launch(void* const* d_in, const int* in_sizes, int n_in,
                              void* d_out, int out_size, void* d_ws, size_t ws_size,
                              hipStream_t stream) {
  (void)in_sizes; (void)n_in; (void)out_size; (void)ws_size;
  const float* x    = (const float*)d_in[0];
  const float* cs   = (const float*)d_in[1];
  const float* sn   = (const float*)d_in[2];
  // d_in[3] = mask (unused; causal mask applied analytically)
  const float* ln1  = (const float*)d_in[4];
  const float* ln2  = (const float*)d_in[5];
  const float* qkvw = (const float*)d_in[6];
  const float* ow   = (const float*)d_in[7];
  const float* guw  = (const float*)d_in[8];
  const float* dww  = (const float*)d_in[9];
  float* h = (float*)d_out;   // residual stream lives in the output buffer

  char* ws = (char*)d_ws;
  size_t off = 0;
  auto carve = [&](size_t bytes) -> void* {
    void* p = ws + off;
    off += (bytes + 255) & ~(size_t)255;
    return p;
  };
  _Float16* y16  = (_Float16*)carve((size_t)LSEQ * DMODEL * 2);
  float*    bigf = (float*)carve((size_t)LSEQ * (2 * IFF) * 4); // qkv then gu
  _Float16* q16  = (_Float16*)carve((size_t)NHEAD * LSEQ * HDIM * 2);
  _Float16* k16  = (_Float16*)carve((size_t)NHEAD * LSEQ * HDIM * 2);
  _Float16* v16  = (_Float16*)carve((size_t)NHEAD * LSEQ * HDIM * 2);
  _Float16* o16  = (_Float16*)carve((size_t)LSEQ * OHH * 2);
  _Float16* a16  = (_Float16*)carve((size_t)LSEQ * IFF * 2);

  copy_f32<<<(LSEQ * DMODEL + 255) / 256, 256, 0, stream>>>(x, h, LSEQ * DMODEL);

  for (int i = 0; i < NLAYER; ++i) {
    const float* qkvw_i = qkvw + (size_t)i * DMODEL * QKVN;
    const float* ow_i   = ow   + (size_t)i * OHH * DMODEL;
    const float* guw_i  = guw  + (size_t)i * DMODEL * (2 * IFF);
    const float* dww_i  = dww  + (size_t)i * IFF * DMODEL;

    // ---- attention ----
    rmsnorm_f16<<<LSEQ, 256, 0, stream>>>(h, ln1 + (size_t)i * DMODEL, y16);
    gemm_wmma<0><<<dim3(LSEQ / BM, QKVN / BN), 256, 0, stream>>>(
        y16, qkvw_i, bigf, LSEQ, QKVN, DMODEL);
    rope_split<<<(LSEQ * NHEAD * (HDIM / 2) + 255) / 256, 256, 0, stream>>>(
        bigf, cs, sn, q16, k16, v16);
    attn_flash<<<dim3(LSEQ / 16, NHEAD), 32, 0, stream>>>(q16, k16, v16, o16);
    gemm_wmma<1><<<dim3(LSEQ / BM, DMODEL / BN), 256, 0, stream>>>(
        o16, ow_i, h, LSEQ, DMODEL, OHH);

    // ---- SwiGLU MLP ----
    rmsnorm_f16<<<LSEQ, 256, 0, stream>>>(h, ln2 + (size_t)i * DMODEL, y16);
    gemm_wmma<0><<<dim3(LSEQ / BM, (2 * IFF) / BN), 256, 0, stream>>>(
        y16, guw_i, bigf, LSEQ, 2 * IFF, DMODEL);
    silu_gate<<<(LSEQ * IFF + 255) / 256, 256, 0, stream>>>(bigf, a16);
    gemm_wmma<1><<<dim3(LSEQ / BM, DMODEL / BN), 256, 0, stream>>>(
        a16, dww_i, h, LSEQ, DMODEL, IFF);
  }
}
